// CRF_15685220565612
// MI455X (gfx1250) — matrix operand order repaired and checked by reading, not compiled, not run on hardware
//
#include <hip/hip_runtime.h>

// ---------------- problem constants (match reference) ----------------
constexpr int B_ALL = 128;
constexpr int S     = 512;
constexpr int T     = 256;          // TAGSET + 2
constexpr int START = T - 2;        // 254
constexpr int STOP  = T - 1;        // 255

// tiling
constexpr int BT      = 16;         // batch rows per workgroup
constexpr int WAVES   = 16;         // one wave per 16-wide column tile
constexpr int THREADS = WAVES * 32; // 512 threads, wave32

// padded LDS strides (avoid 64-bank conflicts; keep rows 8B/4B aligned)
constexpr int ES = T + 8;           // expTransT stride (bf16 elements, even)
constexpr int AS = T + 8;           // A-buffer stride (bf16 elements, even)
constexpr int PS = T + 4;           // part stride (f32 elements, even -> 8B rows)

constexpr size_t SMEM_BYTES =
    (size_t)T * ES * sizeof(__bf16)   // expTrans^T  [cur][prev]  (~135 KB)
  + (size_t)BT * PS * sizeof(float)   // part state  (~16.6 KB)
  + (size_t)BT * AS * sizeof(__bf16)  // bf16 exp(part - m) A buffer (~8.4 KB)
  + 64;                               // per-row max

typedef __attribute__((ext_vector_type(16))) __bf16 v16bf;
typedef __attribute__((ext_vector_type(8)))  __bf16 v8bf;
typedef __attribute__((ext_vector_type(8)))  float  v8f;

static __device__ __forceinline__ unsigned bfbits(float f) {
    unsigned u = __builtin_bit_cast(unsigned, f);
    return (u + 0x7FFFu + ((u >> 16) & 1u)) >> 16;     // RNE to bf16 bits
}
static __device__ __forceinline__ unsigned pack2bf(float lo, float hi) {
    return bfbits(lo) | (bfbits(hi) << 16);            // packed b32 store payload
}

// Workgroup barrier that only drains DS traffic (dscnt), NOT in-flight global
// loads: __syncthreads() lowers to s_wait_loadcnt_dscnt 0x0 + barrier, which
// would stall the step's early feats/mask prefetch on the critical path.
// DS ops complete in order per wave, so dscnt==0 makes all prior LDS
// loads/stores of this wave visible/complete before signaling.
static __device__ __forceinline__ void lds_barrier() {
    asm volatile("s_wait_dscnt 0x0\n\t"
                 "s_barrier_signal -1\n\t"
                 "s_barrier_wait -1" ::: "memory");
}

// ---------------------------------------------------------------------
// Forward algorithm: each workgroup owns a 16-row batch tile and scans
// t = 1..S-1 serially.  Per step: part <- log( exp(part-m) @ expTrans ) + m + emit
// realized as a 16x256 @ 256x256 bf16 WMMA GEMM in exp space.
// ---------------------------------------------------------------------
__global__ __launch_bounds__(THREADS, 1)
void crf_forward_kernel(const float* __restrict__ feats,
                        const float* __restrict__ trans,
                        const int*   __restrict__ mask,
                        float*       __restrict__ wsF) {
    extern __shared__ char smem[];
    __bf16*   expT  = (__bf16*)smem;                                    // [T][ES]
    float*    part  = (float*)(smem + (size_t)T * ES * sizeof(__bf16)); // [BT][PS]
    __bf16*   abuf  = (__bf16*)((char*)part + (size_t)BT * PS * sizeof(float)); // [BT][AS]
    unsigned* abufU = (unsigned*)abuf;
    float*    mrow  = (float*)((char*)abuf + (size_t)BT * AS * sizeof(__bf16)); // [BT]

    const int tid  = threadIdx.x;
    const int lane = tid & 31;
    const int wave = tid >> 5;            // 0..15 : row (steps a/b) & col tile (GEMM)
    const int b0   = blockIdx.x * BT;

    // ---- precompute expTrans^T in bf16 (packed b32 stores):
    //      expT[cur][prev] = exp(trans[prev][cur])
    for (int i = tid; i < (T * T) / 2; i += THREADS) {
        int cur = i >> 7, pr2 = (i & 127) * 2;
        float lo = __expf(trans[pr2 * T + cur]);
        float hi = __expf(trans[(pr2 + 1) * T + cur]);
        *(unsigned*)(expT + cur * ES + pr2) = pack2bf(lo, hi);
    }
    // ---- part0 = emit[t=0] + trans[START][:]
    for (int i = tid; i < BT * T; i += THREADS) {
        int r = i >> 8, c = i & 255;
        part[r * PS + c] = feats[(b0 + r) * S * T + c] + trans[START * T + c];
    }
    __syncthreads();

    const int half = lane >> 4;           // 0: lanes 0-15, 1: lanes 16-31
    const int l16  = lane & 15;
    const int n0   = wave * 16;           // this wave's output column tile
    const int col  = n0 + l16;            // C/D layout: N = lane%16

    const __bf16* Arow = abuf + l16 * AS;          // A: M = lane%16
    const __bf16* Brow = expT + (n0 + l16) * ES;   // B: N = lane%16 (expT rows)

    for (int t = 1; t < S; ++t) {
        // ---- prefetch this step's globals early; they only need to complete
        //      by step (d), i.e. after the whole reduction + GEMM window.
        float emitv[8];
        int   mk[8];
        #pragma unroll
        for (int i = 0; i < 8; ++i) {
            int b = b0 + i + 8 * half;                 // C layout: M = vgpr + 8*(lane/16)
            mk[i]    = mask[b * S + t];
            emitv[i] = feats[(b * S + t) * T + col];
        }

        // (a) per-row max over previous tags: wave w reduces row w.
        //     Column pairs per lane -> 4x ds_load_b64, values kept in registers.
        float pv[8];
        #pragma unroll
        for (int j = 0; j < 4; ++j) {
            float2 p2 = *(const float2*)(part + wave * PS + 2 * lane + 64 * j);
            pv[2 * j] = p2.x; pv[2 * j + 1] = p2.y;
        }
        float mv = -1e30f;
        #pragma unroll
        for (int j = 0; j < 8; ++j) mv = fmaxf(mv, pv[j]);
        #pragma unroll
        for (int off = 16; off > 0; off >>= 1)
            mv = fmaxf(mv, __shfl_xor(mv, off, 32));
        if (lane == 0) mrow[wave] = mv;

        // (b) A = bf16(exp(part - m)), packed pair stores (4x ds_store_b32)
        #pragma unroll
        for (int j = 0; j < 4; ++j) {
            int c = 2 * lane + 64 * j;
            abufU[(wave * AS + c) >> 1] =
                pack2bf(__expf(pv[2 * j] - mv), __expf(pv[2 * j + 1] - mv));
        }
        lds_barrier();   // abuf/mrow visible; global prefetch stays in flight

        // hoist mrow reads so they overlap the GEMM's DS traffic
        float mreg[8];
        #pragma unroll
        for (int i = 0; i < 8; ++i) mreg[i] = mrow[i + 8 * half];

        // (c) 16x16 tile GEMM over K=256 with v_wmma_f32_16x16x32_bf16
        v8f acc = {0.f, 0.f, 0.f, 0.f, 0.f, 0.f, 0.f, 0.f};
        #pragma unroll
        for (int k0 = 0; k0 < T; k0 += 32) {
            // A 16x32 bf16: half-wave 0 holds K {0..7,16..23}, half 1 {8..15,24..31}
            int ka = k0 + half * 8;
            v8bf alo = *(const v8bf*)(Arow + ka);
            v8bf ahi = *(const v8bf*)(Arow + ka + 16);
            // B 32x16 bf16: half-wave 0 holds K {0..7,16..23}, half 1 {8..15,24..31}
            int kb = k0 + half * 8;
            v8bf blo = *(const v8bf*)(Brow + kb);
            v8bf bhi = *(const v8bf*)(Brow + kb + 16);
            v16bf av = __builtin_shufflevector(alo, ahi, 0,1,2,3,4,5,6,7,8,9,10,11,12,13,14,15);
            v16bf bv = __builtin_shufflevector(blo, bhi, 0,1,2,3,4,5,6,7,8,9,10,11,12,13,14,15);
            acc = __builtin_amdgcn_wmma_f32_16x16x32_bf16(
                false, av, false, bv, (short)0, acc, false, false);
        }

        // (d/e) newp = log(s) + m + emit ; masked freeze, write back
        #pragma unroll
        for (int i = 0; i < 8; ++i) {
            if (mk[i] > 0) {
                int row = i + 8 * half;
                part[row * PS + col] = __logf(acc[i]) + mreg[i] + emitv[i];
            }
        }
        lds_barrier();   // part updates visible before next step's reads
    }

    // ---- final: wsF[b] = LSE_prev( part[b][prev] + trans[prev][STOP] )
    {
        float fv[8];
        #pragma unroll
        for (int j = 0; j < 8; ++j) {
            int c = lane + 32 * j;
            fv[j] = part[wave * PS + c] + trans[c * T + STOP];
        }
        float mv = -1e30f;
        #pragma unroll
        for (int j = 0; j < 8; ++j) mv = fmaxf(mv, fv[j]);
        #pragma unroll
        for (int off = 16; off > 0; off >>= 1)
            mv = fmaxf(mv, __shfl_xor(mv, off, 32));
        float sv = 0.f;
        #pragma unroll
        for (int j = 0; j < 8; ++j) sv += __expf(fv[j] - mv);
        #pragma unroll
        for (int off = 16; off > 0; off >>= 1)
            sv += __shfl_xor(sv, off, 32);
        if (lane == 0) wsF[b0 + wave] = mv + __logf(sv);
    }
}

// ---------------- gold path score: one thread per batch row ----------------
__global__ void crf_gold_kernel(const float* __restrict__ feats,
                                const float* __restrict__ trans,
                                const int*   __restrict__ tags,
                                const int*   __restrict__ mask,
                                float*       __restrict__ wsG) {
    int b = blockIdx.x * blockDim.x + threadIdx.x;
    if (b >= B_ALL) return;
    int prev = START;
    float acc = 0.f;
    for (int t = 0; t < S; ++t) {
        if (mask[b * S + t] > 0) {
            int tg = tags[b * S + t];
            acc += feats[(b * S + t) * T + tg] + trans[prev * T + tg];
            prev = tg;                      // prefix mask -> prev ends at end_id
        }
    }
    acc += trans[prev * T + STOP];
    wsG[b] = acc;
}

// ---------------- deterministic fixed-order final reduction ----------------
__global__ void crf_reduce_kernel(const float* __restrict__ wsF,
                                  const float* __restrict__ wsG,
                                  float* __restrict__ out) {
    if (blockIdx.x == 0 && threadIdx.x == 0) {
        float s = 0.f;
        for (int b = 0; b < B_ALL; ++b) s += wsF[b];
        for (int b = 0; b < B_ALL; ++b) s -= wsG[b];
        out[0] = s;
    }
}

extern "C" void kernel_launch(void* const* d_in, const int* in_sizes, int n_in,
                              void* d_out, int out_size, void* d_ws, size_t ws_size,
                              hipStream_t stream) {
    const float* feats = (const float*)d_in[0];
    const float* trans = (const float*)d_in[1];
    const int*   tags  = (const int*)  d_in[2];
    const int*   mask  = (const int*)  d_in[3];
    float* out = (float*)d_out;
    float* wsF = (float*)d_ws;          // 128 floats: per-batch forward scores
    float* wsG = wsF + B_ALL;           // 128 floats: per-batch gold scores

    // >64KB dynamic LDS (CDNA5: 320KB/WGP) — raise the cap every call (deterministic).
    (void)hipFuncSetAttribute(reinterpret_cast<const void*>(crf_forward_kernel),
                              hipFuncAttributeMaxDynamicSharedMemorySize,
                              (int)SMEM_BYTES);

    crf_forward_kernel<<<dim3(B_ALL / BT), dim3(THREADS), SMEM_BYTES, stream>>>(
        feats, trans, mask, wsF);
    crf_gold_kernel<<<dim3(1), dim3(B_ALL), 0, stream>>>(feats, trans, tags, mask, wsG);
    crf_reduce_kernel<<<dim3(1), dim3(32), 0, stream>>>(wsF, wsG, out);
}
